// Attention_45286135169415
// MI455X (gfx1250) — matrix operand order repaired and checked by reading, not compiled, not run on hardware
//
#include <hip/hip_runtime.h>

typedef __attribute__((ext_vector_type(16))) __bf16 v16bf;
typedef __attribute__((ext_vector_type(8)))  float  v8f;

#define UNITS 512
#define DIN   512
#define BATCH 8
#define SEQ   2048

__device__ __forceinline__ unsigned short f32_to_bf16(float f) {
    union { float f; unsigned u; } x; x.f = f;
    unsigned u = x.u;
    u += 0x7FFFu + ((u >> 16) & 1u);   // round-to-nearest-even
    return (unsigned short)(u >> 16);
}

// packed f32x2 -> bf16x2 (v_cvt_pk_bf16_f32 when available)
#if __has_builtin(__builtin_amdgcn_cvt_pk_bf16_f32)
typedef __attribute__((ext_vector_type(2))) __bf16 v2bf;
__device__ __forceinline__ unsigned pack_bf16(float x, float y) {
    union { v2bf v; unsigned u; } c;
    c.v = __builtin_amdgcn_cvt_pk_bf16_f32(x, y);
    return c.u;
}
#else
__device__ __forceinline__ unsigned pack_bf16(float x, float y) {
    return (unsigned)f32_to_bf16(x) | ((unsigned)f32_to_bf16(y) << 16);
}
#endif

union Frag {
    v16bf v;
    uint4 q[2];
    unsigned u[8];
};

// LDS byte address (offset from wave LDS base) for async-to-LDS VDST operand
__device__ __forceinline__ unsigned lds_addr_of(void* p) {
    return (unsigned)(unsigned long long)(__attribute__((address_space(3))) void*)p;
}

// CDNA5 async global->LDS copy, 16 bytes per lane (tracked by ASYNCcnt)
__device__ __forceinline__ void async_ld_b128(unsigned ldsaddr, const void* gaddr) {
    asm volatile("global_load_async_to_lds_b128 %0, %1, off"
                 :: "v"(ldsaddr), "v"(gaddr) : "memory");
}
__device__ __forceinline__ void wait_async_all() {
    asm volatile("s_wait_asynccnt 0" ::: "memory");
}

// ---------------------------------------------------------------------------
// Kernel 1: Q/K/V projections.  C[M=B*S, N=512] = X[M,512] @ W[512,512]
// Block: 256 threads (8 waves), 64x64 tile, waves wm(0..3) x wn(0..1).
// z==2 (V) is stored TRANSPOSED per batch: Vt[b][u][s], so the attention
// kernel can build its P@V B-fragments with contiguous ds_load_b128.
// ---------------------------------------------------------------------------
__global__ void proj_kernel(const float* __restrict__ query,
                            const float* __restrict__ value,
                            const float* __restrict__ W1,
                            const float* __restrict__ W2,
                            const float* __restrict__ W3,
                            unsigned short* __restrict__ Qb,
                            unsigned short* __restrict__ Kb,
                            unsigned short* __restrict__ Vb) {
    __shared__ unsigned short Xs[64 * 32];   // [row][k]
    __shared__ unsigned short Wt[64 * 32];   // TRANSPOSED: [n][k]

    const int tid  = threadIdx.x;
    const int lane = tid & 31;
    const int wave = tid >> 5;
    const int wm   = wave >> 1;    // 0..3  (16-row tile)
    const int wn   = wave & 1;     // 0..1  (32-col tile)
    const int half = lane >> 4;
    const int l16  = lane & 15;

    const float* X = (blockIdx.z == 0) ? query : value;
    const float* W = (blockIdx.z == 0) ? W1 : (blockIdx.z == 1 ? W2 : W3);

    const int row0 = blockIdx.y * 64;
    const int col0 = blockIdx.x * 64;

    v8f acc[2];
    acc[0] = (v8f){}; acc[1] = (v8f){};

    for (int kb = 0; kb < DIN; kb += 32) {
        __syncthreads();
        // stage X tile 64x32 (f32 -> bf16), packed pair stores
        for (int i = tid; i < 64 * 16; i += 256) {
            int r = i >> 4, c2 = (i & 15) * 2;
            float2 f = *(const float2*)&X[(size_t)(row0 + r) * DIN + kb + c2];
            *(unsigned*)&Xs[r * 32 + c2] = pack_bf16(f.x, f.y);
        }
        // stage W tile 32x64 transposed -> Wt[n][k]; pair of k per thread
        for (int i = tid; i < 32 * 32; i += 256) {
            int k2 = (i >> 6) * 2, n = i & 63;
            float a0 = W[(size_t)(kb + k2)     * UNITS + col0 + n];
            float a1 = W[(size_t)(kb + k2 + 1) * UNITS + col0 + n];
            *(unsigned*)&Wt[n * 32 + k2] = pack_bf16(a0, a1);
        }
        __syncthreads();

        // A fragment (two b128 loads per lane)
        Frag a;
        {
            const unsigned short* xb = &Xs[(wm * 16 + l16) * 32];
            a.q[0] = *(const uint4*)(xb + 16 * half);
            a.q[1] = *(const uint4*)(xb + 16 * half + 8);
        }
        // B fragments (two contiguous b128 loads per lane)
        #pragma unroll
        for (int nt = 0; nt < 2; ++nt) {
            Frag b;
            const unsigned short* wb = &Wt[(wn * 32 + nt * 16 + l16) * 32 + 16 * half];
            b.q[0] = *(const uint4*)(wb);
            b.q[1] = *(const uint4*)(wb + 8);
            acc[nt] = __builtin_amdgcn_wmma_f32_16x16x32_bf16(
                false, a.v, false, b.v, (short)0, acc[nt], false, false);
        }
    }

    // store bf16 result: C layout row = v + 8*half, col = lane%16
    if (blockIdx.z == 2) {
        // V transposed: Vt[b][u][s]
        #pragma unroll
        for (int nt = 0; nt < 2; ++nt) {
            #pragma unroll
            for (int v = 0; v < 8; ++v) {
                int row = row0 + wm * 16 + v + 8 * half;     // flattened b*SEQ + s
                int col = col0 + wn * 32 + nt * 16 + l16;    // u
                int bb = row >> 11, s = row & (SEQ - 1);
                Vb[((size_t)bb * UNITS + col) * SEQ + s] = f32_to_bf16(acc[nt][v]);
            }
        }
    } else {
        unsigned short* Out = (blockIdx.z == 0) ? Qb : Kb;
        #pragma unroll
        for (int nt = 0; nt < 2; ++nt) {
            #pragma unroll
            for (int v = 0; v < 8; ++v) {
                int row = row0 + wm * 16 + v + 8 * half;
                int col = col0 + wn * 32 + nt * 16 + l16;
                Out[(size_t)row * UNITS + col] = f32_to_bf16(acc[nt][v]);
            }
        }
    }
}

// ---------------------------------------------------------------------------
// Kernel 2: flash attention.  Block = 256 threads (8 waves) = wm(0..1) x wn(0..3).
// 32 query rows per block.  Key loop in tiles of 32 keys, DOUBLE-BUFFERED:
// tile j+1 is staged with global_load_async_to_lds_b128 (ASYNCcnt) while the
// WMMA work on tile j runs; s_wait_asynccnt 0 at the loop top is the only
// staging stall.  Score work is split across the 4 wn-waves (128 K-dims each),
// partials reduced in LDS by wn==0 which owns the online softmax and
// broadcasts P (bf16) and per-row alpha.  wn selects the 128-col U-chunk of O.
// ---------------------------------------------------------------------------
#define QS_OFF 0
#define KS_OFF (32 * 512 * 2)                       // 2 buffers of 32KB
#define VS_OFF (KS_OFF + 2 * 32 * 512 * 2)          // 2 buffers of 32KB
#define SP_OFF (VS_OFF + 2 * 32 * 512 * 2)          // partials: 2*4*2*256 f32
#define PS_OFF (SP_OFF + 4096 * 4)                  // P: 2 x 16 x 32 bf16
#define AL_OFF (PS_OFF + 2 * 16 * 32 * 2)           // alpha/invl: 32 f32
#define SMEM_BYTES (AL_OFF + 32 * 4)                // 182400 bytes (< 320KB/WGP)

__global__ void attn_kernel(const unsigned short* __restrict__ Qb,
                            const unsigned short* __restrict__ Kb,
                            const unsigned short* __restrict__ Vt,
                            float* __restrict__ out) {
    extern __shared__ char smem[];
    unsigned short* Qs  = (unsigned short*)(smem + QS_OFF);  // [32 q-rows][512]
    unsigned short* Ks0 = (unsigned short*)(smem + KS_OFF);  // [2][32 keys][512]
    unsigned short* Vs0 = (unsigned short*)(smem + VS_OFF);  // [2][512 u][32 keys] (transposed)
    float*          Sp  = (float*)(smem + SP_OFF);           // [wm][wn][nt][row][col]
    unsigned short* Ps  = (unsigned short*)(smem + PS_OFF);  // [wm][16][32]
    float*          Al  = (float*)(smem + AL_OFF);           // [wm][16]

    const int tid  = threadIdx.x;
    const int lane = tid & 31;
    const int wave = tid >> 5;
    const int wm   = wave >> 2;   // 0..1 query row tile
    const int wn   = wave & 3;    // 0..3 (score K-chunk / O U-chunk)
    const int half = lane >> 4;
    const int l16  = lane & 15;

    const int b  = blockIdx.y;
    const int q0 = blockIdx.x * 32;

    const unsigned short* Qg  = Qb + ((size_t)b * SEQ + q0) * UNITS;
    const unsigned short* Kg  = Kb + (size_t)b * SEQ * UNITS;
    const unsigned short* Vtg = Vt + (size_t)b * UNITS * SEQ;

    const unsigned ksBase = lds_addr_of(Ks0);
    const unsigned vsBase = lds_addr_of(Vs0);

    // issue all async copies for key-tile starting at j0 into buffer bufI
    auto stage_tile = [&](int j0, int bufI) {
        const unsigned kb_ = ksBase + (unsigned)bufI * (32 * 512 * 2);
        const unsigned vb_ = vsBase + (unsigned)bufI * (32 * 512 * 2);
        for (int i = tid; i < 2048; i += 256) {        // K: [32 keys][512], 2048 x b128
            int key = i >> 6, q = i & 63;
            async_ld_b128(kb_ + (unsigned)(key * 512 + q * 8) * 2,
                          Kg + (size_t)(j0 + key) * UNITS + q * 8);
        }
        for (int i = tid; i < 2048; i += 256) {        // V^T: [512 u][32 keys], 2048 x b128
            int col = i >> 2, q = i & 3;
            async_ld_b128(vb_ + (unsigned)(col * 32 + q * 8) * 2,
                          Vtg + (size_t)col * SEQ + j0 + q * 8);
        }
    };

    // stage Q tile once (32x512 bf16) as b128 copies; prefetch first key tile
    for (int i = tid; i < 2048; i += 256)
        ((uint4*)Qs)[i] = ((const uint4*)Qg)[i];
    stage_tile(0, 0);

    float mrow[8], lrow[8];
    v8f O[8];
    #pragma unroll
    for (int v = 0; v < 8; ++v) { mrow[v] = -1e30f; lrow[v] = 0.0f; O[v] = (v8f){}; }

    const float scale = 0.044194173824159216f;  // 1/sqrt(512)

    for (int jt = 0; jt < SEQ / 32; ++jt) {
        const int bufI = jt & 1;
        wait_async_all();
        __syncthreads();  // staged tile bufI visible to all waves

        // kick off DMA for the next tile into the other buffer (overlaps compute)
        if (jt + 1 < SEQ / 32) stage_tile((jt + 1) * 32, bufI ^ 1);

        const unsigned short* Ksr = Ks0 + bufI * (32 * 512);
        const unsigned short* Vsr = Vs0 + bufI * (32 * 512);

        // ---- partial scores: this wave covers K-dims [kk*32, kk*32+32) for kk = wn + 4*kk4 ----
        v8f s0 = (v8f){}, s1 = (v8f){};
        const unsigned short* qrow = &Qs[(wm * 16 + l16) * 512];
        #pragma unroll
        for (int kk4 = 0; kk4 < 4; ++kk4) {
            const int kk = wn + kk4 * 4;
            Frag a;
            a.q[0] = *(const uint4*)(qrow + kk * 32 + 16 * half);
            a.q[1] = *(const uint4*)(qrow + kk * 32 + 16 * half + 8);
            Frag b0, b1;
            const unsigned short* k0p = &Ksr[(l16)      * 512 + kk * 32 + 16 * half];
            const unsigned short* k1p = &Ksr[(16 + l16) * 512 + kk * 32 + 16 * half];
            b0.q[0] = *(const uint4*)(k0p); b0.q[1] = *(const uint4*)(k0p + 8);
            b1.q[0] = *(const uint4*)(k1p); b1.q[1] = *(const uint4*)(k1p + 8);
            s0 = __builtin_amdgcn_wmma_f32_16x16x32_bf16(false, a.v, false, b0.v,
                                                         (short)0, s0, false, false);
            s1 = __builtin_amdgcn_wmma_f32_16x16x32_bf16(false, a.v, false, b1.v,
                                                         (short)0, s1, false, false);
        }

        // ---- cross-wave reduction of partials via LDS ----
        if (wn != 0) {
            float* dst = &Sp[((wm * 4 + wn) * 2) * 256];
            #pragma unroll
            for (int v = 0; v < 8; ++v) {
                dst[(v + 8 * half) * 16 + l16]       = s0[v];
                dst[256 + (v + 8 * half) * 16 + l16] = s1[v];
            }
        }
        __syncthreads();

        if (wn == 0) {
            #pragma unroll
            for (int v = 0; v < 8; ++v) {
                int idx = (v + 8 * half) * 16 + l16;
                #pragma unroll
                for (int w = 1; w < 4; ++w) {
                    const float* src = &Sp[((wm * 4 + w) * 2) * 256];
                    s0[v] += src[idx];
                    s1[v] += src[256 + idx];
                }
            }
            // ---- online softmax (row = v + 8*half, cols across 16-lane half) ----
            unsigned short* P = &Ps[wm * 16 * 32];
            #pragma unroll
            for (int v = 0; v < 8; ++v) {
                float x0 = s0[v] * scale, x1 = s1[v] * scale;
                float rmax = fmaxf(x0, x1);
                #pragma unroll
                for (int msk = 1; msk < 16; msk <<= 1)
                    rmax = fmaxf(rmax, __shfl_xor(rmax, msk, 32));
                float mn = fmaxf(mrow[v], rmax);
                float al = __expf(mrow[v] - mn);
                mrow[v] = mn;
                float p0 = __expf(x0 - mn);
                float p1 = __expf(x1 - mn);
                float rs = p0 + p1;
                #pragma unroll
                for (int msk = 1; msk < 16; msk <<= 1)
                    rs += __shfl_xor(rs, msk, 32);
                lrow[v] = lrow[v] * al + rs;
                int row = v + 8 * half;
                P[row * 32 + l16]      = f32_to_bf16(p0);
                P[row * 32 + 16 + l16] = f32_to_bf16(p1);
                Al[wm * 16 + row] = al;   // same value from all 16 lanes of half
            }
        }
        __syncthreads();  // P + alpha visible

        // ---- O = O*alpha + P(16x32) @ V_j(32 x 128-col chunk) ----
        float av[8];
        #pragma unroll
        for (int v = 0; v < 8; ++v) av[v] = Al[wm * 16 + v + 8 * half];

        Frag pa;
        {
            const unsigned short* pb = &Ps[wm * 16 * 32 + l16 * 32];
            pa.q[0] = *(const uint4*)(pb + 16 * half);
            pa.q[1] = *(const uint4*)(pb + 16 * half + 8);
        }
        #pragma unroll
        for (int nt = 0; nt < 8; ++nt) {
            #pragma unroll
            for (int v = 0; v < 8; ++v) O[nt][v] *= av[v];
            Frag bv;
            const unsigned short* vb = &Vsr[(wn * 128 + nt * 16 + l16) * 32 + 16 * half];
            bv.q[0] = *(const uint4*)(vb);
            bv.q[1] = *(const uint4*)(vb + 8);
            O[nt] = __builtin_amdgcn_wmma_f32_16x16x32_bf16(
                false, pa.v, false, bv.v, (short)0, O[nt], false, false);
        }
    }

    // ---- epilogue: broadcast 1/l, normalize, store f32 ----
    __syncthreads();
    if (wn == 0) {
        #pragma unroll
        for (int v = 0; v < 8; ++v)
            Al[wm * 16 + v + 8 * half] = 1.0f / lrow[v];
    }
    __syncthreads();
    float invl[8];
    #pragma unroll
    for (int v = 0; v < 8; ++v) invl[v] = Al[wm * 16 + v + 8 * half];

    float* outp = out + ((size_t)b * SEQ + q0) * UNITS;
    #pragma unroll
    for (int nt = 0; nt < 8; ++nt) {
        #pragma unroll
        for (int v = 0; v < 8; ++v) {
            int row = wm * 16 + v + 8 * half;
            int col = wn * 128 + nt * 16 + l16;
            outp[(size_t)row * UNITS + col] = O[nt][v] * invl[v];
        }
    }
}

// ---------------------------------------------------------------------------
extern "C" void kernel_launch(void* const* d_in, const int* in_sizes, int n_in,
                              void* d_out, int out_size, void* d_ws, size_t ws_size,
                              hipStream_t stream) {
    const float* query = (const float*)d_in[0];
    const float* value = (const float*)d_in[1];
    const float* W1    = (const float*)d_in[2];
    const float* W2    = (const float*)d_in[3];
    const float* W3    = (const float*)d_in[4];

    unsigned short* Qb = (unsigned short*)d_ws;
    unsigned short* Kb = Qb + (size_t)BATCH * SEQ * UNITS;
    unsigned short* Vb = Kb + (size_t)BATCH * SEQ * UNITS;   // transposed [b][u][s]

    // allow >64KB dynamic LDS for the attention kernel (graph-capture safe)
    (void)hipFuncSetAttribute((const void*)attn_kernel,
                              hipFuncAttributeMaxDynamicSharedMemorySize,
                              SMEM_BYTES);

    dim3 pgrid(UNITS / 64, (BATCH * SEQ) / 64, 3);
    proj_kernel<<<pgrid, 256, 0, stream>>>(query, value, W1, W2, W3, Qb, Kb, Vb);

    dim3 agrid(SEQ / 32, BATCH, 1);
    attn_kernel<<<agrid, 256, SMEM_BYTES, stream>>>(Qb, Kb, Vb, (float*)d_out);
}
